// QuantizedLinear_11871289606999
// MI455X (gfx1250) — compile-verified, matrix operand order checked by reading, not run on hardware
//
#include <hip/hip_runtime.h>

typedef __attribute__((ext_vector_type(16))) _Float16 v16h;
typedef __attribute__((ext_vector_type(8)))  _Float16 v8h;
typedef __attribute__((ext_vector_type(8)))  float    v8f;
typedef __attribute__((ext_vector_type(4)))  int      v4i;
typedef __attribute__((ext_vector_type(8)))  int      v8i;

#define IN_F    4096
#define OUT_F   11008
#define MROWS   32
#define GS      128
#define NG      32            // IN_F / GS
#define PK_COLS (IN_F / 2)    // int32 words per packed row (one byte per int32)
#define KSPLIT  4
#define NG_PART (NG / KSPLIT) // 8 groups (1024 K) per split
#define SLAB    (MROWS * OUT_F)   // 352256 f32 per split-K slab

// One wave computes a 32(M) x 16(N) x 1024(K) partial tile:
//   acc0 -> rows 0..15, acc1 -> rows 16..31, both sharing one dequantized B tile.
// Grid: (11008/64) x KSPLIT = 172 x 4 = 688 blocks of 4 waves -> 2752 waves.
// The 8-group x 4-step K loop is fully unrolled so ALL loads are
// base-pointer + compile-time immediate: no in-loop address math, and the
// scheduler can issue the weight stream several K-steps ahead of the WMMAs.
__global__ __launch_bounds__(128) void QuantizedLinear_w4a16_wmma(
    const _Float16* __restrict__ x,        // [32, 4096] fp16
    const int*      __restrict__ packed,   // [11008, 2048] int32, one byte each (2 nibbles)
    const _Float16* __restrict__ scales,   // [11008, 32] fp16
    const _Float16* __restrict__ zeros,    // [11008, 32] fp16
    float*          __restrict__ ws)       // [KSPLIT, 32, 11008] f32 partials
{
    const int lane = threadIdx.x & 31;
    const int wave = threadIdx.x >> 5;
    const int l15  = lane & 15;
    const int hi   = lane >> 4;                      // lane group: 0 = lanes 0-15, 1 = lanes 16-31
    const int kp   = blockIdx.y;                     // split-K part
    const int g0   = kp * NG_PART;                   // first quant group of this part

    const int n = blockIdx.x * 64 + wave * 16 + l15; // output column == W row held by this lane

    // Per-part base pointers (computed once; everything below is immediate offsets).
    const int*      pq  = packed + (size_t)n * PK_COLS + (size_t)g0 * (GS / 2) + hi * 4;
    const _Float16* px0 = x + (size_t)l15 * IN_F        + (size_t)g0 * GS + hi * 8;
    const _Float16* px1 = x + (size_t)(16 + l15) * IN_F + (size_t)g0 * GS + hi * 8;
    const _Float16* ps  = scales + (size_t)n * NG + g0;
    const _Float16* pz  = zeros  + (size_t)n * NG + g0;

    v8f acc0 = {};
    v8f acc1 = {};

    #pragma unroll
    for (int g = 0; g < NG_PART; ++g) {
        // w = q*s + z, with q' = (1024+q) materialized by bit-OR into the fp16
        // mantissa (0x6400 | q == half(1024+q)).  Fold the -1024*s into z once:
        //   w = q' * s + (z - 1024*s)
        const _Float16 s  = ps[g];
        const _Float16 z  = pz[g];
        const _Float16 zp = z - (_Float16)1024.0f * s;
        v16h sv, zv;
        #pragma unroll
        for (int i = 0; i < 16; ++i) { sv[i] = s; zv[i] = zp; }

        #pragma unroll
        for (int kk = 0; kk < 4; ++kk) {
            const int xo = g * GS + kk * 32;      // half offset (immediate)
            const int qo = g * (GS / 2) + kk * 16; // word offset (immediate)

            // ---- A operands: x tiles, ISA 16-bit A-matrix layout (two b128 loads each) ----
            v8h a_lo0 = *(const v8h*)(px0 + xo);
            v8h a_hi0 = *(const v8h*)(px0 + xo + 16);
            v8h a_lo1 = *(const v8h*)(px1 + xo);
            v8h a_hi1 = *(const v8h*)(px1 + xo + 16);
            v16h A0 = __builtin_shufflevector(a_lo0, a_hi0,
                        0,1,2,3,4,5,6,7,8,9,10,11,12,13,14,15);
            v16h A1 = __builtin_shufflevector(a_lo1, a_hi1,
                        0,1,2,3,4,5,6,7,8,9,10,11,12,13,14,15);

            // ---- B operand: stream int4 weights (non-temporal), magic-number dequant ----
            // word q (value 0..255): element 2c (even K) = high nibble -> low half,
            //                        element 2c+1 (odd K) = low nibble -> high half.
            //   ((q<<16)|(q>>4)) & 0x000f000f | 0x64006400  -> v_lshl_or + v_and_or
            v4i qa = __builtin_nontemporal_load((const v4i*)(pq + qo));
            v4i qb = __builtin_nontemporal_load((const v4i*)(pq + qo + 8));
            v8i bb;
            #pragma unroll
            for (int c = 0; c < 4; ++c) {
                const unsigned qlo = (unsigned)qa[c];   // < 256 by construction
                const unsigned qhi = (unsigned)qb[c];
                bb[c]     = (int)((((qlo << 16) | (qlo >> 4)) & 0x000f000fu) | 0x64006400u);
                bb[4 + c] = (int)((((qhi << 16) | (qhi >> 4)) & 0x000f000fu) | 0x64006400u);
            }
            v16h Bq = __builtin_bit_cast(v16h, bb);     // halves = 1024 + nibble
            v16h B  = Bq * sv + zv;                     // -> v_pk_fma_f16 x8

            // D = A x B + C  (v_wmma_f32_16x16x32_f16), B shared by both M tiles
            acc0 = __builtin_amdgcn_wmma_f32_16x16x32_f16(
                       false, A0, false, B, (short)0, acc0, false, false);
            acc1 = __builtin_amdgcn_wmma_f32_16x16x32_f16(
                       false, A1, false, B, (short)0, acc1, false, false);
        }
    }

    // ---- epilogue: store f32 partials to this split's slab (non-temporal) ----
    // f32 C/D layout: VGPR r holds row (hi*8 + r) for this lane's half; column = n.
    float* wsl = ws + (size_t)kp * SLAB + (size_t)hi * 8 * OUT_F + n;
    #pragma unroll
    for (int r = 0; r < 8; ++r) {
        __builtin_nontemporal_store(acc0[r], wsl + (size_t)r * OUT_F);
        __builtin_nontemporal_store(acc1[r], wsl + (size_t)(16 + r) * OUT_F);
    }
}

// Sum the KSPLIT partial slabs + bias, convert to fp16.  32*11008 = 352256
// elements = 1376 blocks x 256 threads exactly.
__global__ __launch_bounds__(256) void QuantizedLinear_reduce(
    const float*    __restrict__ ws,
    const _Float16* __restrict__ bias,
    _Float16*       __restrict__ out)
{
    const int idx = blockIdx.x * 256 + threadIdx.x;   // m * OUT_F + n
    const int n   = idx % OUT_F;
    float v = ws[idx] + ws[idx + SLAB] + ws[idx + 2 * SLAB] + ws[idx + 3 * SLAB];
    out[idx] = (_Float16)(v + (float)bias[n]);
}

extern "C" void kernel_launch(void* const* d_in, const int* in_sizes, int n_in,
                              void* d_out, int out_size, void* d_ws, size_t ws_size,
                              hipStream_t stream) {
    (void)in_sizes; (void)n_in; (void)out_size; (void)ws_size;
    const _Float16* x      = (const _Float16*)d_in[0];
    const int*      packed = (const int*)     d_in[1];
    const _Float16* scales = (const _Float16*)d_in[2];
    const _Float16* zeros  = (const _Float16*)d_in[3];
    const _Float16* bias   = (const _Float16*)d_in[4];
    // d_in[5] = group_size (128), d_in[6] = in_features (4096): compile-time constants here.
    _Float16* out = (_Float16*)d_out;
    float*    ws  = (float*)d_ws;      // needs KSPLIT*32*11008*4 B ~= 5.4 MiB

    dim3 grid(OUT_F / 64, KSPLIT);     // 172 x 4 = 688 blocks
    dim3 block(128);                   // 4 waves
    hipLaunchKernelGGL(QuantizedLinear_w4a16_wmma, grid, block, 0, stream,
                       x, packed, scales, zeros, ws);

    dim3 rgrid((MROWS * OUT_F) / 256); // 1376 blocks
    hipLaunchKernelGGL(QuantizedLinear_reduce, rgrid, dim3(256), 0, stream,
                       ws, bias, out);
}
